// ScaledDotProductAttention_4758823764164
// MI455X (gfx1250) — matrix operand order repaired and checked by reading, not compiled, not run on hardware
//
#include <hip/hip_runtime.h>

// CDNA5 / gfx1250 sparse top-k attention, round 5.
//  Kernel A (x3): convert Q/K/V f32 -> bf16 into d_ws (nontemporal f32 reads).
//  Kernel B: per block = 16 query rows of one batch, 8 wave32 waves.
//   Pass 1: S = (Q K^T)/sqrt(D) via v_wmma_f32_16x16x32_bf16. K tiles streamed
//           into LDS double-buffered: TDM tensor_load_to_lds (TENSORcnt) when
//           available, else global_load_async_to_lds_b128 (ASYNCcnt).
//   Pass 2: softmax + iterative top-16 per row (wave32 shuffles), sparsify,
//           renormalize; sparse_attn -> HBM (nontemporal), bf16 probs -> LDS.
//   Pass 3: O = sparse_attn @ V, V chunks double-buffered async into LDS,
//           A-fragments read directly as bf16 (no per-element convert).

typedef __attribute__((ext_vector_type(16))) __bf16 v16bf;
typedef __attribute__((ext_vector_type(8)))  float  v8f;
typedef int      v4i __attribute__((ext_vector_type(4)));
typedef int      v8i __attribute__((ext_vector_type(8)));
typedef unsigned v4u __attribute__((ext_vector_type(4)));

#define B_N 64
#define LQn 1024
#define LKn 1024
#define DD  128
#define TOPKN 16
#define NTHREADS 256
#define TENSOR_ELEMS ((size_t)B_N * LKn * DD)   // 8,388,608 per tensor

#if __has_builtin(__builtin_amdgcn_global_load_async_to_lds_b128)
#define HAVE_ASYNC 1
#else
#define HAVE_ASYNC 0
#endif

#if __has_builtin(__builtin_amdgcn_tensor_load_to_lds)
#define HAVE_TDM 1
#else
#define HAVE_TDM 0
#endif

// Builtin expects AS1 ("__device__") int4* source and AS3 int4* dest.
typedef __attribute__((address_space(1))) v4i gv4i;
typedef __attribute__((address_space(3))) v4i lv4i;

// 16-byte global -> LDS copy, async (ASYNCcnt) when available.
__device__ __forceinline__ void async_cp16(const void* g, void* l) {
#if HAVE_ASYNC
  __builtin_amdgcn_global_load_async_to_lds_b128(
      (gv4i*)(unsigned long long)g,
      (lv4i*)(unsigned)(unsigned long long)l, 0, 0);
#else
  *(uint4*)l = *(const uint4*)g;
#endif
}

template <int N>
__device__ __forceinline__ void wait_async() {
#if HAVE_ASYNC
#if __has_builtin(__builtin_amdgcn_s_wait_asynccnt)
  __builtin_amdgcn_s_wait_asynccnt(N);
#else
  asm volatile("s_wait_asynccnt %0" :: "n"(N) : "memory");
#endif
#endif
}

#if HAVE_TDM
// One-shot 1-D TDM copy: n contiguous 2-byte elements, global -> LDS.
// D# per CDNA5 ISA ch.8: group0 {count=1, lds_addr, global_addr, type=2},
// group1 {data_size=2B, tensor_dim0=n, tile_dim0=n}, groups 2/3 zero (<=2D).
// This toolchain exposes the 6-arg builtin (extra int32x8 before cpol).
__device__ __forceinline__ void tdm_load_1d_u16(const void* gsrc, void* lds, int n) {
  unsigned long long ga = (unsigned long long)gsrc;
  unsigned la = (unsigned)(unsigned long long)lds;
  v4u g0;
  g0[0] = 1u;                                   // count=1, user mode
  g0[1] = la;                                   // lds_addr [63:32]
  g0[2] = (unsigned)ga;                         // global_addr [95:64]
  g0[3] = ((unsigned)(ga >> 32) & 0x01FFFFFFu)  // global_addr [120:96]
        | 0x80000000u;                          // type=2 ("image") [127:126]
  v8i g1 = {};
  g1[0] = 0x10000;                              // data_size=1 -> 2 bytes
  g1[1] = (int)(((unsigned)n & 0xFFFFu) << 16); // tensor_dim0 low16 @ [63:48]
  g1[2] = (int)((unsigned)n >> 16);             // tensor_dim0 high16 @ [79:64]
  g1[3] = (int)(((unsigned)n & 0xFFFFu) << 16); // tile_dim0 @ [127:112]
  v4i z4 = {};
  v8i z8 = {};
  __builtin_amdgcn_tensor_load_to_lds(g0, g1, z4, z4, z8, 0);
}
#endif

// Wait until at most PENDING K-stages are still in flight.
template <int PENDING>
__device__ __forceinline__ void wait_k_stages() {
#if HAVE_TDM
#if __has_builtin(__builtin_amdgcn_s_wait_tensorcnt)
  __builtin_amdgcn_s_wait_tensorcnt(PENDING);
#else
  asm volatile("s_wait_tensorcnt %0" :: "n"(PENDING) : "memory");
#endif
#else
  wait_async<PENDING * 8>();
#endif
}

union FragBF { v16bf v; unsigned short s[16]; };

__device__ __forceinline__ unsigned short f2bf(float f) {
  union { float f; unsigned u; } c; c.f = f;
  unsigned u = c.u;
  return (unsigned short)((u + 0x7FFFu + ((u >> 16) & 1u)) >> 16);   // RNE
}

// A-matrix (16x32 bf16, MxK): K index for vector element e, lane-half h.
__device__ __forceinline__ int kmapA(int e, int h) {
  int vv = e >> 1, j = e & 1;
  return ((vv & 3) << 1) + j + (h << 3) + ((vv >> 2) << 4);
}
// B-matrix (32x16 bf16, KxN): K index for vector element e, lane-half h.
__device__ __forceinline__ int kmapB(int e, int h) {
  int vv = e >> 1, j = e & 1;
  return ((vv & 3) << 1) + j + (h << 3) + ((vv >> 2) << 4);
}

// ---------------- f32 -> bf16 pre-pass (memory bound) ----------------
__global__ __launch_bounds__(NTHREADS)
void cvt_bf16_kernel(const float* __restrict__ src, unsigned short* __restrict__ dst) {
  size_t base = ((size_t)blockIdx.x * NTHREADS + threadIdx.x) * 8;
  unsigned p[8];
  #pragma unroll
  for (int i = 0; i < 8; ++i) {
    float f = __builtin_nontemporal_load(src + base + i);  // don't pollute L2
    union { float f; unsigned u; } c; c.f = f;
    p[i] = (c.u + 0x7FFFu + ((c.u >> 16) & 1u)) >> 16;
  }
  uint4 o;
  o.x = p[0] | (p[1] << 16);
  o.y = p[2] | (p[3] << 16);
  o.z = p[4] | (p[5] << 16);
  o.w = p[6] | (p[7] << 16);
  *(uint4*)(dst + base) = o;      // regular store: bf16 stays L2-resident
}

// ---------------- main attention kernel ----------------
__global__ __launch_bounds__(NTHREADS)
void sparse_topk_attn_kernel(const unsigned short* __restrict__ qb16,
                             const unsigned short* __restrict__ kb16,
                             const unsigned short* __restrict__ vb16,
                             float* __restrict__ out,
                             float* __restrict__ out_attn)
{
  extern __shared__ char smem_raw[];
  float*          s_scores = (float*)smem_raw;                           // [16][1024] f32 (64 KB)
  unsigned short* s_q      = (unsigned short*)(smem_raw + 65536);        // [16][128] bf16 ( 4 KB)
  unsigned short* s_k      = (unsigned short*)(smem_raw + 65536 + 4096); // 2x[128][128] bf16 (64 KB)
  unsigned short* s_v      = s_k;                                        // 2x[32][128] bf16 (16 KB alias)
  unsigned short* s_p      = s_k + 24576;  // [16][1024] bf16 (32 KB), after V buffers

  const int tid  = threadIdx.x;
  const int wave = tid >> 5;
  const int lane = tid & 31;
  const int hl   = lane >> 4;
  const int ln16 = lane & 15;

  const int blk = blockIdx.x;
  const int b   = blk >> 6;
  const int q0  = (blk & 63) << 4;

  const float invT = 0.08838834764831845f; // 1/sqrt(128)

  const unsigned short* qg = qb16 + ((size_t)b * LQn + q0) * DD;
  const unsigned short* kg = kb16 + (size_t)b * LKn * DD;
  const unsigned short* vg = vb16 + (size_t)b * LKn * DD;

  // ---- stage Q tile (4 KB async) + K group 0 (32 KB) ----
  async_cp16((const char*)qg + tid * 16, (char*)s_q + tid * 16);
#if HAVE_TDM
  if (wave == 0)
    tdm_load_1d_u16(kg, s_k, 16384);
#else
  #pragma unroll
  for (int i = 0; i < 8; ++i)
    async_cp16((const char*)kg + tid * 16 + i * 4096,
               (char*)s_k + tid * 16 + i * 4096);
#endif

  // ---- Pass 1: scores, double-buffered K staging ----
  for (int grp = 0; grp < 8; ++grp) {
    const int cur = grp & 1;
    if (grp < 7) {
#if HAVE_TDM
      if (wave == 0)
        tdm_load_1d_u16((const char*)kg + (size_t)(grp + 1) * 32768,
                        (char*)s_k + (cur ^ 1) * 32768, 16384);
#else
      const char* src = (const char*)kg + (size_t)(grp + 1) * 32768;
      char*       dst = (char*)s_k + (cur ^ 1) * 32768;
      #pragma unroll
      for (int i = 0; i < 8; ++i)
        async_cp16(src + tid * 16 + i * 4096, dst + tid * 16 + i * 4096);
#endif
      wait_k_stages<1>();   // only the stage just issued may remain in flight
    } else {
      wait_k_stages<0>();
    }
#if HAVE_TDM
    if (grp == 0) wait_async<0>();   // Q tile (ASYNCcnt) ready
#endif
    __syncthreads();

    const unsigned short* kbuf = s_k + cur * 16384;
    v8f acc = {};
    #pragma unroll
    for (int dc = 0; dc < 4; ++dc) {
      const int d0 = dc * 32;
      FragBF a, bfr;
      #pragma unroll
      for (int e = 0; e < 16; ++e) {
        a.s[e]   = s_q[ln16 * DD + d0 + kmapA(e, hl)];
        bfr.s[e] = kbuf[(wave * 16 + ln16) * DD + d0 + kmapB(e, hl)];
      }
      acc = __builtin_amdgcn_wmma_f32_16x16x32_bf16(
          false, a.v, false, bfr.v, (short)0, acc, false, false);
    }
    const int colbase = (grp * 8 + wave) * 16 + ln16;
    #pragma unroll
    for (int vv = 0; vv < 8; ++vv)
      s_scores[(vv + 8 * hl) * LKn + colbase] = acc[vv] * invT;
    __syncthreads();   // buffer cur fully consumed before it is re-staged
  }

  // prefetch V chunk 0 (8 KB) during the ALU-heavy pass 2
  #pragma unroll
  for (int i = 0; i < 2; ++i)
    async_cp16((const char*)vg + tid * 16 + i * 4096,
               (char*)s_v + tid * 16 + i * 4096);

  // ---- Pass 2: softmax + top-16 + renormalize (2 rows per wave) ----
  float* attn_rowbase = out_attn + ((size_t)b * LQn + q0) * (size_t)LKn;
  #pragma unroll 1
  for (int rr = 0; rr < 2; ++rr) {
    const int row = wave * 2 + rr;
    float valv[32];
    #pragma unroll
    for (int i = 0; i < 32; ++i) valv[i] = s_scores[row * LKn + lane + 32 * i];

    float m = valv[0];
    #pragma unroll
    for (int i = 1; i < 32; ++i) m = fmaxf(m, valv[i]);
    #pragma unroll
    for (int off = 16; off > 0; off >>= 1) m = fmaxf(m, __shfl_xor(m, off, 32));

    float ssum = 0.f;
    #pragma unroll
    for (int i = 0; i < 32; ++i) { valv[i] = __expf(valv[i] - m); ssum += valv[i]; }
    #pragma unroll
    for (int off = 16; off > 0; off >>= 1) ssum += __shfl_xor(ssum, off, 32);
    const float inv_s = 1.0f / ssum;
    #pragma unroll
    for (int i = 0; i < 32; ++i) valv[i] *= inv_s;   // attn probabilities

    // iterative selection of 16 largest (exact with duplicates)
    float sel[32];
    #pragma unroll
    for (int i = 0; i < 32; ++i) sel[i] = valv[i];
    float kth = 0.f;
    #pragma unroll 1
    for (int it = 0; it < TOPKN; ++it) {
      float lv = sel[0]; int li = 0;
      #pragma unroll
      for (int i = 1; i < 32; ++i) { if (sel[i] > lv) { lv = sel[i]; li = i; } }
      float bv = lv; int bl = lane;
      #pragma unroll
      for (int off = 16; off > 0; off >>= 1) {
        float ov = __shfl_xor(bv, off, 32);
        int   ol = __shfl_xor(bl, off, 32);
        if (ov > bv || (ov == bv && ol < bl)) { bv = ov; bl = ol; }
      }
      if (lane == bl) {
        #pragma unroll
        for (int i = 0; i < 32; ++i) { if (i == li) sel[i] = -1.0f; }
      }
      kth = bv;
    }

    const float thr = kth + 1e-8f;
    float ws = 0.f;
    #pragma unroll
    for (int i = 0; i < 32; ++i) { valv[i] = fmaxf(valv[i] - thr, 0.f); ws += valv[i]; }
    #pragma unroll
    for (int off = 16; off > 0; off >>= 1) ws += __shfl_xor(ws, off, 32);
    const float invw = 1.0f / (ws + 1e-8f);

    float* grow = attn_rowbase + (size_t)row * LKn;
    #pragma unroll
    for (int i = 0; i < 32; ++i) {
      float sp = valv[i] * invw;
      s_p[row * LKn + lane + 32 * i] = f2bf(sp);             // bf16 for PV pass
      __builtin_nontemporal_store(sp, grow + lane + 32 * i); // 268 MB stream: keep out of L2
    }
  }
  __syncthreads();

  // ---- Pass 3: O = sparse_attn @ V, double-buffered V staging ----
  v8f oacc = {};
  for (int kc = 0; kc < 32; ++kc) {
    const int cur = kc & 1;
    if (kc < 31) {
      const char* src = (const char*)vg + (size_t)(kc + 1) * 8192;
      char*       dst = (char*)s_v + (cur ^ 1) * 8192;
      async_cp16(src + tid * 16,        dst + tid * 16);
      async_cp16(src + tid * 16 + 4096, dst + tid * 16 + 4096);
      wait_async<2>();
    } else {
      wait_async<0>();
    }
    __syncthreads();

    const unsigned short* vbuf = s_v + cur * 4096;
    FragBF a, bfr;
    #pragma unroll
    for (int e = 0; e < 16; ++e) {
      a.s[e]   = s_p[ln16 * LKn + kc * 32 + kmapA(e, hl)];
      bfr.s[e] = vbuf[kmapB(e, hl) * DD + wave * 16 + ln16];
    }
    oacc = __builtin_amdgcn_wmma_f32_16x16x32_bf16(
        false, a.v, false, bfr.v, (short)0, oacc, false, false);
    __syncthreads();
  }

  float* ob = out + ((size_t)b * LQn + q0) * DD + wave * 16 + ln16;
  #pragma unroll
  for (int vv = 0; vv < 8; ++vv)
    __builtin_nontemporal_store(oacc[vv], ob + (size_t)(vv + 8 * hl) * DD);
}

extern "C" void kernel_launch(void* const* d_in, const int* in_sizes, int n_in,
                              void* d_out, int out_size, void* d_ws, size_t ws_size,
                              hipStream_t stream) {
  (void)in_sizes; (void)n_in; (void)out_size; (void)ws_size;
  const float* q = (const float*)d_in[0];
  const float* k = (const float*)d_in[1];
  const float* v = (const float*)d_in[2];
  float* out      = (float*)d_out;
  float* out_attn = out + (size_t)B_N * LQn * DD;

  unsigned short* qb16 = (unsigned short*)d_ws;
  unsigned short* kb16 = qb16 + TENSOR_ELEMS;
  unsigned short* vb16 = kb16 + TENSOR_ELEMS;

  const int cvt_blocks = (int)(TENSOR_ELEMS / 8 / NTHREADS);   // 4096
  cvt_bf16_kernel<<<cvt_blocks, NTHREADS, 0, stream>>>(q, qb16);
  cvt_bf16_kernel<<<cvt_blocks, NTHREADS, 0, stream>>>(k, kb16);
  cvt_bf16_kernel<<<cvt_blocks, NTHREADS, 0, stream>>>(v, vb16);

  const size_t smem = 65536 + 4096 + 65536;   // 132 KB of the WGP's 320 KB LDS
  (void)hipFuncSetAttribute((const void*)sparse_topk_attn_kernel,
                            hipFuncAttributeMaxDynamicSharedMemorySize, (int)smem);
  sparse_topk_attn_kernel<<<B_N * (LQn / 16), NTHREADS, smem, stream>>>(
      qb16, kb16, vb16, out, out_attn);
}
